// FP8PerTensorQKVLinear_22179211117323
// MI455X (gfx1250) — 8-bit matrix operand order repaired and checked by reading, compiled, not run on hardware
//
#include <hip/hip_runtime.h>
#include <stdint.h>

typedef __attribute__((ext_vector_type(16))) int   v16i;
typedef __attribute__((ext_vector_type(8)))  float v8f;

#define M_DIM 16384
#define K_DIM 4096
#define N_DIM 6144
#define DQ    4096
#define DKV   1024
#define FP8_MAX 448.0f

__device__ __forceinline__ unsigned int f32_to_e4m3_sw(float f) {
    unsigned int u = __float_as_uint(f);
    unsigned int sign = (u >> 24) & 0x80u;
    float a = fabsf(f);
    a = fminf(a, FP8_MAX);
    if (a < 0.0009765625f) return sign;
    int e;
    float m = frexpf(a, &e);
    int exp = e - 1 + 7;
    unsigned int bits;
    if (exp >= 1) {
        int q = (int)rintf(m * 16.0f);
        if (q == 16) { q = 8; exp += 1; }
        if (exp > 15) { exp = 15; q = 14; }
        bits = ((unsigned)exp << 3) | ((unsigned)q & 7u);
    } else {
        int q = (int)rintf(a * 512.0f);
        bits = (q > 7) ? 0x08u : (unsigned)q;
    }
    return sign | bits;
}

__global__ __launch_bounds__(256)
void quant_fp8_kernel(const float* __restrict__ src,
                      unsigned int* __restrict__ dst,
                      const float* __restrict__ s_ptr,
                      int n4) {
    int i = blockIdx.x * 256 + threadIdx.x;
    if (i >= n4) return;
    float inv = 1.0f;
    if (s_ptr) inv = 1.0f / s_ptr[0];
    const float4 v = ((const float4*)src)[i];
    float a0 = fminf(fmaxf(v.x * inv, -FP8_MAX), FP8_MAX);
    float a1 = fminf(fmaxf(v.y * inv, -FP8_MAX), FP8_MAX);
    float a2 = fminf(fmaxf(v.z * inv, -FP8_MAX), FP8_MAX);
    float a3 = fminf(fmaxf(v.w * inv, -FP8_MAX), FP8_MAX);
#if __has_builtin(__builtin_amdgcn_cvt_pk_fp8_f32)
    int p = __builtin_amdgcn_cvt_pk_fp8_f32(a0, a1, 0, false);
    p     = __builtin_amdgcn_cvt_pk_fp8_f32(a2, a3, p, true);
    dst[i] = (unsigned int)p;
#else
    unsigned int p = f32_to_e4m3_sw(a0)
                   | (f32_to_e4m3_sw(a1) << 8)
                   | (f32_to_e4m3_sw(a2) << 16)
                   | (f32_to_e4m3_sw(a3) << 24);
    dst[i] = p;
#endif
}

__global__ __launch_bounds__(256)
void qkv_gemm_fp8_kernel(const unsigned char* __restrict__ Aq,
                         const unsigned char* __restrict__ Wq,
                         float* __restrict__ out,
                         const float* __restrict__ sq,
                         const float* __restrict__ sk,
                         const float* __restrict__ sv,
                         const float* __restrict__ sin_) {
    const int lane = threadIdx.x & 31;
    const int wave = threadIdx.x >> 5;
    const int unit = __builtin_amdgcn_readfirstlane(blockIdx.x * 8 + wave);
    const int NT   = N_DIM / 64;
    const int m_tile = unit / NT;
    const int n_grp  = unit % NT;

    const int r    = lane & 15;
    const int half = lane >> 4;

    const unsigned char* arow =
        Aq + (size_t)(m_tile * 16 + r) * K_DIM + 16 * half;

    v8f acc[4] = {};

    for (int k = 0; k < K_DIM; k += 128) {
        v16i a;
#pragma unroll
        for (int g = 0; g < 8; ++g) {
            uint64_t q = *(const uint64_t*)(arow + k + 32 * (g >> 1) + 8 * (g & 1));
            a[2 * g]     = (int)(uint32_t)q;
            a[2 * g + 1] = (int)(uint32_t)(q >> 32);
        }
        __builtin_prefetch(arow + k + 128, 0, 3);

#pragma unroll
        for (int t = 0; t < 4; ++t) {
            const int col = n_grp * 64 + t * 16 + r;
            const unsigned char* bp =
                Wq + (size_t)col * K_DIM + k + 16 * half;
            v16i b;
#pragma unroll
            for (int q2 = 0; q2 < 4; ++q2) {
                const uint4 u = *(const uint4*)(bp + 32 * q2);
                b[4 * q2 + 0] = (int)u.x;
                b[4 * q2 + 1] = (int)u.y;
                b[4 * q2 + 2] = (int)u.z;
                b[4 * q2 + 3] = (int)u.w;
            }
            __builtin_prefetch(bp + 128, 0, 3);
            acc[t] = __builtin_amdgcn_wmma_f32_16x16x128_fp8_fp8(
                a, b, (short)0, acc[t], false, false);
        }
    }

    const float s_in = sin_[0];
#pragma unroll
    for (int t = 0; t < 4; ++t) {
        const int n0 = n_grp * 64 + t * 16;
        const float* sp = (n0 < DQ) ? sq : ((n0 < DQ + DKV) ? sk : sv);
        const float scale = s_in * sp[0];
#pragma unroll
        for (int v = 0; v < 8; ++v) {
            const int row = m_tile * 16 + v + 8 * half;
            out[(size_t)row * N_DIM + n0 + r] = acc[t][v] * scale;
        }
    }
}

extern "C" void kernel_launch(void* const* d_in, const int* in_sizes, int n_in,
                              void* d_out, int out_size, void* d_ws, size_t ws_size,
                              hipStream_t stream) {
    const float* x    = (const float*)d_in[0];
    const float* wq   = (const float*)d_in[1];
    const float* wk   = (const float*)d_in[2];
    const float* wv   = (const float*)d_in[3];
    const float* s_q  = (const float*)d_in[4];
    const float* s_k  = (const float*)d_in[5];
    const float* s_v  = (const float*)d_in[6];
    const float* s_in = (const float*)d_in[7];

    unsigned char* xq   = (unsigned char*)d_ws;
    unsigned char* wfp8 = xq + (size_t)M_DIM * K_DIM;

    {
        int n4 = M_DIM * (K_DIM / 4);
        quant_fp8_kernel<<<(n4 + 255) / 256, 256, 0, stream>>>(
            x, (unsigned int*)xq, s_in, n4);
    }
    {
        int n4q  = DQ  * (K_DIM / 4);
        int n4kv = DKV * (K_DIM / 4);
        quant_fp8_kernel<<<(n4q + 255) / 256, 256, 0, stream>>>(
            wq, (unsigned int*)wfp8, nullptr, n4q);
        quant_fp8_kernel<<<(n4kv + 255) / 256, 256, 0, stream>>>(
            wk, (unsigned int*)(wfp8 + (size_t)DQ * K_DIM), nullptr, n4kv);
        quant_fp8_kernel<<<(n4kv + 255) / 256, 256, 0, stream>>>(
            wv, (unsigned int*)(wfp8 + (size_t)(DQ + DKV) * K_DIM), nullptr, n4kv);
    }
    {
        const int units  = (M_DIM / 16) * (N_DIM / 64);
        const int blocks = units / 8;
        qkv_gemm_fp8_kernel<<<blocks, 256, 0, stream>>>(
            xq, wfp8, (float*)d_out, s_q, s_k, s_v, s_in);
    }
}
